// TimeSeriesAutoencoder_56203942035887
// MI455X (gfx1250) — compile-verified, matrix operand order checked by reading, not run on hardware
//
#include <hip/hip_runtime.h>
#include <cstdint>
#include <cstddef>

// CDNA5 / gfx1250: wave32, WMMA 16x16x32 bf16 -> f32 accum.
typedef __attribute__((ext_vector_type(16))) __bf16 v16bf;
typedef __attribute__((ext_vector_type(8)))  float  v8f;
typedef __attribute__((ext_vector_type(4)))  unsigned int u32x4;
typedef __attribute__((ext_vector_type(8)))  int  i32x8;
typedef __attribute__((ext_vector_type(4)))  int  i32x4;

union ABf { v16bf v; uint4 q[2]; };

__device__ __forceinline__ float bf2f(unsigned short u) {
  union { unsigned int i; float f; } t; t.i = ((unsigned int)u) << 16; return t.f;
}
__device__ __forceinline__ unsigned short f2bf(float f) {
  union { float f; unsigned int i; } t; t.f = f;
  unsigned int r = t.i + 0x7FFFu + ((t.i >> 16) & 1u);
  return (unsigned short)(r >> 16);
}
__device__ __forceinline__ float sigmf(float x) { return 1.f / (1.f + __expf(-x)); }

// ---------------------------------------------------------------------------
// TDM: 1-D contiguous tile DMA  global -> LDS  (Tensor Data Mover, gfx1250).
// Descriptor per CDNA5 ISA ch.8: group0 {count=1, lds_addr, global_addr, type=2},
// group1 {data_size=8B, tensor_dim0 = tile_dim0 = bytes/8, tensor_dim1=1}.
// Tracked with TENSORcnt; completion via s_wait_tensorcnt.
// ---------------------------------------------------------------------------
__device__ __forceinline__ void tdm_load_1d(unsigned lds_off, const void* gptr, unsigned bytes) {
  unsigned long long ga = (unsigned long long)(uintptr_t)gptr;
  unsigned n8 = bytes >> 3;  // 8-byte units
  u32x4 g0;
  g0.x = 0x1u;                                             // count=1 (valid descriptor)
  g0.y = lds_off;                                          // lds_addr (bytes)
  g0.z = (unsigned)(ga & 0xFFFFFFFFu);                     // global_addr[31:0]
  g0.w = (unsigned)((ga >> 32) & 0x01FFFFFFu) | (2u << 30); // global_addr[56:32] | type=2
  i32x8 g1;
  g1[0] = (int)(3u << 16);                                 // data_size = 8 bytes
  g1[1] = (int)((n8 & 0xFFFFu) << 16);                     // tensor_dim0[15:0]  @ bits 63:48
  g1[2] = (int)(((n8 >> 16) & 0xFFFFu) | (1u << 16));      // tensor_dim0[31:16] | tensor_dim1=1
  g1[3] = (int)((n8 & 0xFFFFu) << 16);                     // tile_dim0 @ bits 127:112
  g1[4] = 0;                                               // tile_dim1=0 (1-D), tile_dim2=0
  g1[5] = (int)n8;                                         // tensor_dim0_stride[31:0]
  g1[6] = 0;
  g1[7] = 0;
  i32x4 gz = {0, 0, 0, 0};
#if defined(__clang_major__) && __clang_major__ >= 23
  i32x8 gz8 = {0, 0, 0, 0, 0, 0, 0, 0};
  __builtin_amdgcn_tensor_load_to_lds(g0, g1, gz, gz, gz8, 0);
#else
  __builtin_amdgcn_tensor_load_to_lds(g0, g1, gz, gz, 0);
#endif
}

// ---------------------------------------------------------------------------
// Elementwise prep kernels
// ---------------------------------------------------------------------------
__global__ void cast_f32_bf16(const float* __restrict__ s, unsigned short* __restrict__ d, int n) {
  int i = blockIdx.x * 256 + threadIdx.x;
  if (i < n) d[i] = f2bf(s[i]);
}

// x[B,S,D] fp32 -> xb[S,B,D] bf16 (time-major for the layer pipeline)
__global__ void transpose_cast_x(const float* __restrict__ x, unsigned short* __restrict__ xb,
                                 int B, int S, int D) {
  int i = blockIdx.x * 256 + threadIdx.x;
  int n = B * S * D;
  if (i >= n) return;
  int d = i % D; int t = i / D; int b = t % B; int s = t / B;
  xb[i] = f2bf(x[((size_t)b * S + s) * D + d]);
}

__global__ void fold_bias(const float* __restrict__ a, const float* __restrict__ b,
                          float* __restrict__ o, int n) {
  int i = blockIdx.x * 256 + threadIdx.x;
  if (i < n) o[i] = a[i] + b[i];
}

// ---------------------------------------------------------------------------
// Generic WMMA GEMM:  C[r, n] = sum_k A[arow(r), k] * W[n, k]  (+ bias[n])
//   A row-major [*, lda] bf16, W row-major [N, K] bf16 (i.e. C = A @ W^T).
//   amode: 0 identity; 1 broadcast (arow = r % B); 2 permute (r=b*S+s -> arow=s*B+b)
//   cmode: 0 store bf16; 1 store fp32. ldc = N.
// One wave computes a 16x64 tile (4 N-tiles sharing the A fragment).
// ---------------------------------------------------------------------------
__global__ void __launch_bounds__(256) wmma_gemm_bf16(
    const unsigned short* __restrict__ A, const unsigned short* __restrict__ W,
    const float* __restrict__ bias, void* __restrict__ Cout,
    int M, int N, int K, int lda, int amode, int S, int B, int cmode)
{
  const int lane = threadIdx.x & 31;
  const int wave = threadIdx.x >> 5;
  const int nt64 = N >> 6;
  const int tid  = blockIdx.x * 8 + wave;
  const int total = (M >> 4) * nt64;
  if (tid >= total) return;
  const int mt = tid / nt64, nt = tid % nt64;
  const int m0 = mt << 4, n0 = nt << 6;
  const int half = lane >> 4, lr = lane & 15;

  int r = m0 + lr;
  int arow;
  if (amode == 1)      arow = r % B;
  else if (amode == 2) { int b_ = r / S; int s_ = r - b_ * S; arow = s_ * B + b_; }
  else                 arow = r;

  const unsigned short* Ap = A + (size_t)arow * lda + half * 8;
  const unsigned short* Wp = W + (size_t)(n0 + lr) * K + half * 8;
  const size_t wstep = (size_t)16 * K;  // 16 W rows per N-tile

  v8f acc0 = {}, acc1 = {}, acc2 = {}, acc3 = {};
  const int ktn = K >> 5;
  for (int kt = 0; kt < ktn; ++kt) {
    ABf a;
    a.q[0] = *(const uint4*)(Ap + kt * 32);
    a.q[1] = *(const uint4*)(Ap + kt * 32 + 16);
    const unsigned short* wp = Wp + kt * 32;
    ABf b0, b1, b2, b3;
    b0.q[0] = *(const uint4*)(wp);             b0.q[1] = *(const uint4*)(wp + 16);
    b1.q[0] = *(const uint4*)(wp + wstep);     b1.q[1] = *(const uint4*)(wp + wstep + 16);
    b2.q[0] = *(const uint4*)(wp + 2 * wstep); b2.q[1] = *(const uint4*)(wp + 2 * wstep + 16);
    b3.q[0] = *(const uint4*)(wp + 3 * wstep); b3.q[1] = *(const uint4*)(wp + 3 * wstep + 16);
    acc0 = __builtin_amdgcn_wmma_f32_16x16x32_bf16(false, a.v, false, b0.v, (short)0, acc0, false, false);
    acc1 = __builtin_amdgcn_wmma_f32_16x16x32_bf16(false, a.v, false, b1.v, (short)0, acc1, false, false);
    acc2 = __builtin_amdgcn_wmma_f32_16x16x32_bf16(false, a.v, false, b2.v, (short)0, acc2, false, false);
    acc3 = __builtin_amdgcn_wmma_f32_16x16x32_bf16(false, a.v, false, b3.v, (short)0, acc3, false, false);
  }

  float bc0 = bias ? bias[n0 + 0 * 16 + lr] : 0.f;
  float bc1 = bias ? bias[n0 + 1 * 16 + lr] : 0.f;
  float bc2 = bias ? bias[n0 + 2 * 16 + lr] : 0.f;
  float bc3 = bias ? bias[n0 + 3 * 16 + lr] : 0.f;
  #pragma unroll
  for (int v = 0; v < 8; ++v) {
    int orow = m0 + v + 8 * half;  // C layout: VGPR v -> M=v (lanes<16), M=v+8 (lanes>=16)
    if (cmode == 0) {
      unsigned short* Cp = (unsigned short*)Cout + (size_t)orow * N + n0 + lr;
      Cp[0]  = f2bf(acc0[v] + bc0);
      Cp[16] = f2bf(acc1[v] + bc1);
      Cp[32] = f2bf(acc2[v] + bc2);
      Cp[48] = f2bf(acc3[v] + bc3);
    } else {
      float* Cp = (float*)Cout + (size_t)orow * N + n0 + lr;
      Cp[0]  = acc0[v] + bc0;
      Cp[16] = acc1[v] + bc1;
      Cp[32] = acc2[v] + bc2;
      Cp[48] = acc3[v] + bc3;
    }
  }
}

// ---------------------------------------------------------------------------
// Persistent LSTM recurrence. H = 512, 4H = 2048 hardcoded.
// Grid: 8 blocks (batch tiles of 16 rows), 1024 threads = 32 waves.
// Wave w owns h-columns [16w, 16w+16) and computes gate tiles at
// {16w, 16w+512, 16w+1024, 16w+1536}: the nonlinearity is wave-local.
// h (bf16) lives in LDS; c lives in lane-private registers (each lane
// exclusively owns (row = v+8*half, col = colL) elements).
// The 64KB per-step Xg gate tile is DMA'd into LDS by the Tensor Data Mover
// (wave 0 issues TENSOR_LOAD_TO_LDS), overlapped with the WMMA phase that
// streams Whh (2MB/step) from L2; wave 0 s_wait_tensorcnt's before the
// barrier that publishes the tile.
// ---------------------------------------------------------------------------
__global__ void __launch_bounds__(1024, 1) lstm_recur(
    const unsigned short* __restrict__ Xg, const unsigned short* __restrict__ Whh,
    const int* __restrict__ lengths,
    const unsigned short* __restrict__ h0, const float* __restrict__ c0,
    unsigned short* __restrict__ ys, float* __restrict__ encOut,
    unsigned short* __restrict__ hDump, float* __restrict__ cDump,
    int S, int B)
{
  __shared__ unsigned short hS[16 * 512];    // 16 KB: h state (A operand)
  __shared__ unsigned short xgS[16 * 2048];  // 64 KB: staged Xg tile for current step
  const int b0 = blockIdx.x * 16;

  const int lane = threadIdx.x & 31;
  const int wave = threadIdx.x >> 5;
  const int half = lane >> 4, lr = lane & 15;
  const int colL = wave * 16 + lr;  // owned h column for this lane

  for (int i = threadIdx.x; i < 16 * 512; i += 1024) {
    int row = i >> 9, col = i & 511;
    hS[i] = h0 ? h0[(size_t)(b0 + row) * 512 + col] : (unsigned short)0;
  }

  int   lenv[8];
  float creg[8];
  #pragma unroll
  for (int v = 0; v < 8; ++v) {
    int row = v + 8 * half;
    lenv[v] = lengths[b0 + row];
    creg[v] = c0 ? c0[(size_t)(b0 + row) * 512 + colL] : 0.f;
  }

  const unsigned xg_lds = (unsigned)(uintptr_t)(void*)xgS;  // LDS aperture: addr[31:0]
  if (threadIdx.x == 0)
    tdm_load_1d(xg_lds, Xg + (size_t)(0 * B + b0) * 2048, 16 * 2048 * 2);
  __syncthreads();

  const unsigned short* Wp = Whh + (size_t)colL * 512 + half * 8;  // gate 0 row base

  for (int s = 0; s < S; ++s) {
    v8f acc0 = {}, acc1 = {}, acc2 = {}, acc3 = {};
    #pragma unroll 2
    for (int kt = 0; kt < 16; ++kt) {
      int abase = lr * 512 + kt * 32 + half * 8;  // 16B-aligned LDS offsets
      ABf a;
      a.q[0] = *(const uint4*)&hS[abase];
      a.q[1] = *(const uint4*)&hS[abase + 16];
      const unsigned short* wp = Wp + kt * 32;
      ABf b0_, b1_, b2_, b3_;
      b0_.q[0] = *(const uint4*)(wp);               b0_.q[1] = *(const uint4*)(wp + 16);
      b1_.q[0] = *(const uint4*)(wp + 262144);      b1_.q[1] = *(const uint4*)(wp + 262144 + 16);
      b2_.q[0] = *(const uint4*)(wp + 524288);      b2_.q[1] = *(const uint4*)(wp + 524288 + 16);
      b3_.q[0] = *(const uint4*)(wp + 786432);      b3_.q[1] = *(const uint4*)(wp + 786432 + 16);
      acc0 = __builtin_amdgcn_wmma_f32_16x16x32_bf16(false, a.v, false, b0_.v, (short)0, acc0, false, false);
      acc1 = __builtin_amdgcn_wmma_f32_16x16x32_bf16(false, a.v, false, b1_.v, (short)0, acc1, false, false);
      acc2 = __builtin_amdgcn_wmma_f32_16x16x32_bf16(false, a.v, false, b2_.v, (short)0, acc2, false, false);
      acc3 = __builtin_amdgcn_wmma_f32_16x16x32_bf16(false, a.v, false, b3_.v, (short)0, acc3, false, false);
    }

    if (threadIdx.x == 0) __builtin_amdgcn_s_wait_tensorcnt(0);  // Xg tile s landed
    __syncthreads();  // publishes xgS; also: all waves done reading hS fragments

    #pragma unroll
    for (int v = 0; v < 8; ++v) {
      int row = v + 8 * half;
      const unsigned short* xgp = xgS + row * 2048;
      float gi = sigmf(acc0[v] + bf2f(xgp[colL]));            // torch gate order i,f,g,o
      float gf = sigmf(acc1[v] + bf2f(xgp[512 + colL]));
      float gg = tanhf(acc2[v] + bf2f(xgp[1024 + colL]));
      float go = sigmf(acc3[v] + bf2f(xgp[1536 + colL]));
      int li = row * 512 + colL;
      float c_new = gf * creg[v] + gi * gg;
      float h_new = go * tanhf(c_new);
      bool m = s < lenv[v];  // packed-sequence semantics: freeze state past length
      creg[v] = m ? c_new : creg[v];
      hS[li] = m ? f2bf(h_new) : hS[li];
      ys[(size_t)(s * B + b0 + row) * 512 + colL] = m ? f2bf(h_new) : (unsigned short)0;
      if (encOut) encOut[((size_t)(b0 + row) * S + s) * 512 + colL] = m ? h_new : 0.f;
    }
    __syncthreads();  // hS updated; xgS consumption complete -> safe to overwrite

    if (threadIdx.x == 0 && s + 1 < S)
      tdm_load_1d(xg_lds, Xg + (size_t)((s + 1) * B + b0) * 2048, 16 * 2048 * 2);
  }

  if (hDump) {
    for (int i = threadIdx.x; i < 16 * 512; i += 1024) {
      int row = i >> 9, col = i & 511;
      hDump[(size_t)(b0 + row) * 512 + col] = hS[i];
    }
    #pragma unroll
    for (int v = 0; v < 8; ++v) {
      int row = v + 8 * half;
      cDump[(size_t)(b0 + row) * 512 + colL] = creg[v];
    }
  }
}

// ---------------------------------------------------------------------------
extern "C" void kernel_launch(void* const* d_in, const int* in_sizes, int n_in,
                              void* d_out, int out_size, void* d_ws, size_t ws_size,
                              hipStream_t stream)
{
  (void)in_sizes; (void)n_in; (void)out_size; (void)ws_size;
  const int B = 128, S = 512, D = 128, H = 512;
  const int G = 4 * H;  // 2048

  const float* x        = (const float*)d_in[0];
  const int*   lengths  = (const int*)d_in[1];
  const float* enc_Wih0 = (const float*)d_in[2];
  const float* enc_Wih  = (const float*)d_in[3];
  const float* enc_Whh  = (const float*)d_in[4];
  const float* enc_bih  = (const float*)d_in[5];
  const float* enc_bhh  = (const float*)d_in[6];
  const float* dec_Wih  = (const float*)d_in[7];
  const float* dec_Whh  = (const float*)d_in[8];
  const float* dec_bih  = (const float*)d_in[9];
  const float* dec_bhh  = (const float*)d_in[10];
  const float* out_W    = (const float*)d_in[11];
  const float* out_b    = (const float*)d_in[12];

  char* ws = (char*)d_ws;
  size_t off = 0;
  auto take = [&](size_t bytes) -> char* {
    char* p = ws + off; off += (bytes + 255) & ~(size_t)255; return p;
  };
  unsigned short* xb      = (unsigned short*)take((size_t)S * B * D * 2);
  unsigned short* w_enc0  = (unsigned short*)take((size_t)G * D * 2);
  unsigned short* w_encih = (unsigned short*)take((size_t)3 * G * H * 2);
  unsigned short* w_enchh = (unsigned short*)take((size_t)4 * G * H * 2);
  unsigned short* w_decih = (unsigned short*)take((size_t)4 * G * H * 2);
  unsigned short* w_dechh = (unsigned short*)take((size_t)4 * G * H * 2);
  unsigned short* w_out   = (unsigned short*)take((size_t)D * H * 2);
  float*          bsum    = (float*)take((size_t)8 * G * 4);
  unsigned short* Xg      = (unsigned short*)take((size_t)S * B * G * 2);
  unsigned short* bufA    = (unsigned short*)take((size_t)S * B * H * 2);
  unsigned short* bufB    = (unsigned short*)take((size_t)S * B * H * 2);
  unsigned short* hsv     = (unsigned short*)take((size_t)4 * B * H * 2);
  float*          csv     = (float*)take((size_t)4 * B * H * 4);

  // ---- prep: casts + bias folds ----
  {
    int n = B * S * D;
    transpose_cast_x<<<(n + 255) / 256, 256, 0, stream>>>(x, xb, B, S, D);
    n = G * D;         cast_f32_bf16<<<(n + 255) / 256, 256, 0, stream>>>(enc_Wih0, w_enc0, n);
    n = 3 * G * H;     cast_f32_bf16<<<(n + 255) / 256, 256, 0, stream>>>(enc_Wih, w_encih, n);
    n = 4 * G * H;     cast_f32_bf16<<<(n + 255) / 256, 256, 0, stream>>>(enc_Whh, w_enchh, n);
    n = 4 * G * H;     cast_f32_bf16<<<(n + 255) / 256, 256, 0, stream>>>(dec_Wih, w_decih, n);
    n = 4 * G * H;     cast_f32_bf16<<<(n + 255) / 256, 256, 0, stream>>>(dec_Whh, w_dechh, n);
    n = D * H;         cast_f32_bf16<<<(n + 255) / 256, 256, 0, stream>>>(out_W, w_out, n);
    n = 4 * G;
    fold_bias<<<(n + 255) / 256, 256, 0, stream>>>(enc_bih, enc_bhh, bsum, n);
    fold_bias<<<(n + 255) / 256, 256, 0, stream>>>(dec_bih, dec_bhh, bsum + 4 * G, n);
  }

  float* out_main = (float*)d_out;                       // [B,S,D]
  float* out_enc  = (float*)d_out + (size_t)B * S * D;   // [B,S,H]

  const int M = S * B;
  auto gemm_blocks = [&](int m, int n) { return ((m / 16) * (n / 64) + 7) / 8; };

  // ---- encoder ----
  unsigned short* cur = xb;
  int curK = D;
  for (int l = 0; l < 4; ++l) {
    const unsigned short* Wih = (l == 0) ? w_enc0 : (w_encih + (size_t)(l - 1) * G * H);
    wmma_gemm_bf16<<<gemm_blocks(M, G), 256, 0, stream>>>(
        cur, Wih, bsum + (size_t)l * G, Xg, M, G, curK, curK, /*amode=*/0, S, B, /*cmode=*/0);
    unsigned short* yout = (l % 2 == 0) ? bufA : bufB;
    lstm_recur<<<8, 1024, 0, stream>>>(
        Xg, w_enchh + (size_t)l * G * H, lengths, nullptr, nullptr,
        yout, (l == 3) ? out_enc : nullptr,
        hsv + (size_t)l * B * H, csv + (size_t)l * B * H, S, B);
    cur = yout;
    curK = H;
  }

  // ---- decoder (teacher_forcing=False: input = broadcast of encoder top h) ----
  for (int l = 0; l < 4; ++l) {
    const unsigned short* Ain = (l == 0) ? (hsv + (size_t)3 * B * H) : cur;
    int amode = (l == 0) ? 1 : 0;
    wmma_gemm_bf16<<<gemm_blocks(M, G), 256, 0, stream>>>(
        Ain, w_decih + (size_t)l * G * H, bsum + (size_t)(4 + l) * G,
        Xg, M, G, H, H, amode, S, B, /*cmode=*/0);
    unsigned short* yout = (l % 2 == 0) ? bufA : bufB;
    lstm_recur<<<8, 1024, 0, stream>>>(
        Xg, w_dechh + (size_t)l * G * H, lengths,
        hsv + (size_t)l * B * H, csv + (size_t)l * B * H,
        yout, nullptr, nullptr, nullptr, S, B);
    cur = yout;
  }

  // ---- output linear: [B*S, H] @ [H, D]^T + out_b -> d_out [B,S,D] (fp32) ----
  wmma_gemm_bf16<<<gemm_blocks(B * S, D), 256, 0, stream>>>(
      cur, w_out, out_b, out_main, B * S, D, H, H, /*amode=*/2, S, B, /*cmode=*/1);
}
